// LinearAttentionLayer_10479720203028
// MI455X (gfx1250) — compile-verified, matrix operand order checked by reading, not run on hardware
//
#include <hip/hip_runtime.h>
#include <math.h>
#include <stdint.h>

// ---------------------------------------------------------------------------
// Problem constants: B=4, T=4096, D=1024, H=16, F=4096, DH=64
// ---------------------------------------------------------------------------
#define BB 4
#define TT 4096
#define DD 1024
#define HH 16
#define FF 4096
#define DH 64
#define MM (BB * TT)   // 16384 rows

typedef __bf16 bf;
typedef __attribute__((ext_vector_type(16))) __bf16 v16bf;
typedef __attribute__((ext_vector_type(8)))  float  v8f;

// ---------------------------------------------------------------------------
// WMMA fragment maps (ISA 7.12.2, wave32, 16x16x32 bf16):
//   A 16x32: lane holds m=lane%16;  k = (e&7) + ((e>>3)<<4) + ((lane>>4)<<3)
//   B 32x16: lane holds n=lane%16;  same k map
//   C 16x16: m = e + 8*(lane>>4);   n = lane%16
// Tensors are stored pre-swizzled so one fragment = 512 contiguous bf16 and
// each lane loads its 32 bytes with two global_load_b128.
// ---------------------------------------------------------------------------
__device__ __forceinline__ int kmap(int e, int lane) {
  return (e & 7) + ((e >> 3) << 4) + ((lane >> 4) << 3);
}
// element address inside an A-swizzled tensor (rows M, cols Kc, Kc%32==0)
__device__ __forceinline__ size_t aswz(int m, int k, int Kc) {
  const int ln = (m & 15) | (((k >> 3) & 1) << 4);
  const int e  = (k & 7) | (((k >> 4) & 1) << 3);
  return ((((size_t)(m >> 4)) * (Kc >> 5) + (k >> 5)) << 9) + ln * 16 + e;
}
// element address inside a B-swizzled tensor (rows K, cols N, N%16==0)
__device__ __forceinline__ size_t bswz(int k, int n, int N) {
  const int ln = (n & 15) | (((k >> 3) & 1) << 4);
  const int e  = (k & 7) | (((k >> 4) & 1) << 3);
  return ((((size_t)(k >> 5)) * (N >> 4) + (n >> 4)) << 9) + ln * 16 + e;
}

__device__ __forceinline__ v8f wmma_bf16(v16bf a, v16bf b, v8f c) {
  return __builtin_amdgcn_wmma_f32_16x16x32_bf16(false, a, false, b, (short)0,
                                                 c, false, false);
}

__device__ __forceinline__ float phi_f(float v) {   // elu(v)+1
  return v > 0.0f ? v + 1.0f : __expf(v);
}
__device__ __forceinline__ float gelu_f(float v) {  // exact gelu
  return 0.5f * v * (1.0f + erff(v * 0.70710678118654752f));
}

// ---------------------------------------------------------------------------
// LDS-free WMMA GEMM on pre-swizzled operands, register double-buffered.
// Block: 256 threads = 8 waves (4 m x 2 n). Block tile 128x128; wave tile
// 32x64 -> acc[2][4]; per 32-k step: 6 fragment loads (12 b128), 8 WMMA,
// with next step's fragments loaded before this step's WMMAs issue.
// EPI: 0 phi->bf16 (Q) | 1 phi*valid->bf16 (K) | 2 valid->bf16 (V)
//      4 gelu->bf16 (FFN1) | 5 residual+v -> f32 row-major (out-proj / FFN2)
// bf16 outputs are written A-swizzled for the next GEMM.
// ---------------------------------------------------------------------------
template <int EPI>
__global__ __launch_bounds__(256) void gemm_swz(
    const bf* __restrict__ A, const bf* __restrict__ W,
    const float* __restrict__ bias, const float* __restrict__ residual,
    const unsigned char* __restrict__ mask, bf* __restrict__ outB,
    float* __restrict__ outF, int M, int N, int K) {
  const int lane = threadIdx.x & 31;
  const int wave = threadIdx.x >> 5;
  const int mw = wave & 3, nw = wave >> 2;
  const int mBase = blockIdx.y * 128 + mw * 32;
  const int nBase = blockIdx.x * 128 + nw * 64;
  const int ktiles = K >> 5;
  const int ntiles = N >> 4;
  const int mt0 = mBase >> 4;
  const int nt0 = nBase >> 4;

  v8f acc[2][4] = {};
  const bf* aBase = A + (((size_t)mt0 * ktiles) << 9) + lane * 16;
  const bf* bBase = W + ((size_t)nt0 << 9) + lane * 16;

  // prologue: fragments for kb = 0
  v16bf a0 = *(const v16bf*)(aBase);
  v16bf a1 = *(const v16bf*)(aBase + ((size_t)ktiles << 9));
  v16bf fb[4];
#pragma unroll
  for (int j = 0; j < 4; ++j)
    fb[j] = *(const v16bf*)(bBase + ((size_t)j << 9));

  for (int kb = 0; kb < ktiles; ++kb) {
    const int nk = (kb + 1 < ktiles) ? kb + 1 : kb;  // redundant reload on last
    v16bf na0 = *(const v16bf*)(aBase + ((size_t)nk << 9));
    v16bf na1 = *(const v16bf*)(aBase + (((size_t)ktiles + nk) << 9));
    const bf* nbp = bBase + (((size_t)nk * ntiles) << 9);
    v16bf nfb[4];
#pragma unroll
    for (int j = 0; j < 4; ++j)
      nfb[j] = *(const v16bf*)(nbp + ((size_t)j << 9));

#pragma unroll
    for (int j = 0; j < 4; ++j) {
      acc[0][j] = wmma_bf16(a0, fb[j], acc[0][j]);
      acc[1][j] = wmma_bf16(a1, fb[j], acc[1][j]);
    }
    a0 = na0;
    a1 = na1;
#pragma unroll
    for (int j = 0; j < 4; ++j) fb[j] = nfb[j];
  }

  const int lhi = (lane >> 4) << 3;
#pragma unroll
  for (int i = 0; i < 2; ++i)
#pragma unroll
    for (int j = 0; j < 4; ++j)
#pragma unroll
      for (int e = 0; e < 8; ++e) {
        const int m = mBase + i * 16 + lhi + e;
        const int n = nBase + j * 16 + (lane & 15);
        float v = acc[i][j][e] + bias[n];
        if constexpr (EPI == 0) {
          outB[aswz(m, n, N)] = (bf)phi_f(v);
        } else if constexpr (EPI == 1) {
          const float fm = mask[m] ? 0.0f : 1.0f;
          outB[aswz(m, n, N)] = (bf)(phi_f(v) * fm);
        } else if constexpr (EPI == 2) {
          const float fm = mask[m] ? 0.0f : 1.0f;
          outB[aswz(m, n, N)] = (bf)(v * fm);
        } else if constexpr (EPI == 4) {
          outB[aswz(m, n, N)] = (bf)gelu_f(v);
        } else {
          outF[(size_t)m * N + n] = residual[(size_t)m * N + n] + v;
        }
      }
}

// ---------------------------------------------------------------------------
// KV[b,h] (64x64) = sum_t K[b,t,h,:]^T V[b,t,h,:]  -- one block per (b,h).
// K/V arrive A-swizzled; each 32-t slab = 4 contiguous 1KB fragments per
// tensor, staged into LDS with GLOBAL_LOAD_ASYNC_TO_LDS_B128 (ASYNCcnt),
// then gathered (transposed for A) per element. Output B-swizzled.
// ---------------------------------------------------------------------------
__device__ __forceinline__ int region_addr(int t, int c) {  // t<32, c<64
  const int f  = ((t >> 4) << 1) | (c >> 5);
  const int ln = (t & 15) | (((c >> 3) & 1) << 4);
  const int e  = (c & 7) | (((c >> 4) & 1) << 3);
  return f * 512 + ln * 16 + e;
}

__global__ __launch_bounds__(256) void kv_kernel(const bf* __restrict__ Kb,
                                                 const bf* __restrict__ Vb,
                                                 bf* __restrict__ kvb) {
  __shared__ __align__(16) bf ldsK[2048];
  __shared__ __align__(16) bf ldsV[2048];
  const int bh = blockIdx.x;
  const int b = bh / HH, h = bh % HH;
  const int tid = threadIdx.x, lane = tid & 31, wave = tid >> 5;
  const int m_tile = wave & 3, n_tile0 = (wave >> 2) * 2;
  const int f = tid >> 6, off = (tid & 63) * 8;  // fragment copy mapping
  const int mi = f >> 1, ki = f & 1;

  // LDS byte addresses for this thread's 16B staging slots (ISA 10.2: the
  // LDS aperture truncates a flat shared address to addr[31:0]).
  const uint32_t ldsAddrK = (uint32_t)(uintptr_t)&ldsK[f * 512 + off];
  const uint32_t ldsAddrV = (uint32_t)(uintptr_t)&ldsV[f * 512 + off];

  v8f acc0 = {};
  v8f acc1 = {};

  for (int t0 = 0; t0 < TT; t0 += 32) {
    const size_t src =
        (((size_t)(((b * TT + t0) >> 4) + mi) * (DD >> 5) + 2 * h + ki) << 9) + off;
    const bf* gK = Kb + src;
    const bf* gV = Vb + src;
    // async memory -> LDS (tracked with ASYNCcnt)
    asm volatile("global_load_async_to_lds_b128 %0, %1, off"
                 :: "v"(ldsAddrK), "v"(gK) : "memory");
    asm volatile("global_load_async_to_lds_b128 %0, %1, off"
                 :: "v"(ldsAddrV), "v"(gV) : "memory");
    asm volatile("s_wait_asynccnt 0x0" ::: "memory");
    __syncthreads();

    v16bf fa, b0, b1;
    const int dA = m_tile * 16 + (lane & 15);
    const int n0 = n_tile0 * 16 + (lane & 15);
#pragma unroll
    for (int e = 0; e < 16; ++e) {
      const int tl = kmap(e, lane);
      fa[e] = ldsK[region_addr(tl, dA)];        // A[m=d][k=t] = K^T
      b0[e] = ldsV[region_addr(tl, n0)];
      b1[e] = ldsV[region_addr(tl, n0 + 16)];
    }
    acc0 = wmma_bf16(fa, b0, acc0);
    acc1 = wmma_bf16(fa, b1, acc1);
    __syncthreads();
  }

  const int mLoc = m_tile * 16 + ((lane >> 4) << 3);
  const int n0 = n_tile0 * 16 + (lane & 15);
  bf* out = kvb + ((size_t)bh << 12);
#pragma unroll
  for (int e = 0; e < 8; ++e) {
    out[bswz(mLoc + e, n0, 64)] = (bf)acc0[e];
    out[bswz(mLoc + e, n0 + 16, 64)] = (bf)acc1[e];
  }
}

// Ksum[b,h,d] = sum_t K[b,t,h,d]   (K is A-swizzled)
__global__ void ksum_kernel(const bf* __restrict__ Kb, float* __restrict__ ksum) {
  __shared__ float red[256];
  const int bh = blockIdx.x;
  const int b = bh / HH, h = bh % HH;
  const int d = threadIdx.x & 63, chunk = threadIdx.x >> 6;
  float s = 0.0f;
  for (int t = chunk; t < TT; t += 4)
    s += (float)Kb[aswz(b * TT + t, h * DH + d, DD)];
  red[threadIdx.x] = s;
  __syncthreads();
  if (threadIdx.x < 64)
    ksum[(size_t)bh * 64 + threadIdx.x] =
        red[threadIdx.x] + red[threadIdx.x + 64] +
        red[threadIdx.x + 128] + red[threadIdx.x + 192];
}

// zinv[(b*T+t)*H + h] = 1 / (Q[b,t,h,:] . Ksum[b,h,:] + 1e-6)  (Q A-swizzled)
__global__ void zinv_kernel(const bf* __restrict__ Qb,
                            const float* __restrict__ ksum,
                            float* __restrict__ zinv) {
  const int idx = blockIdx.x * blockDim.x + threadIdx.x;
  if (idx >= MM * HH) return;
  const int h = idx % HH, r = idx / HH;
  const int b = r / TT;
  const float* ks = ksum + (size_t)(b * HH + h) * DH;
  float s = 0.0f;
#pragma unroll 8
  for (int d = 0; d < DH; ++d)
    s += (float)Qb[aswz(r, h * DH + d, DD)] * ks[d];
  zinv[idx] = 1.0f / (s + 1e-6f);
}

// attnO[b,t,h,:] = (Q[b,t,h,:] @ KV[b,h]) * zinv -- LDS-free, fully swizzled.
// Block: 128 t-rows x one (b,h); wave tile 32 rows x 32 cols.
__global__ __launch_bounds__(256) void attn_out_swz(
    const bf* __restrict__ Qb, const bf* __restrict__ kvb,
    const float* __restrict__ zinv, bf* __restrict__ attnO) {
  const int lane = threadIdx.x & 31, wave = threadIdx.x >> 5;
  const int mw = wave & 3, nw = wave >> 2;
  const int bh = blockIdx.y, b = bh / HH, h = bh % HH;
  const int r0 = b * TT + blockIdx.x * 128 + mw * 32;
  const int mt0 = r0 >> 4;

  v8f acc[2][2] = {};
  const bf* aBase = Qb + (((size_t)mt0 * (DD >> 5)) << 9) + lane * 16;
  const bf* bBase = kvb + ((size_t)bh << 12) + lane * 16;

#pragma unroll
  for (int kb = 0; kb < 2; ++kb) {
    const v16bf a0 = *(const v16bf*)(aBase + (((size_t)2 * h + kb) << 9));
    const v16bf a1 =
        *(const v16bf*)(aBase + (((size_t)(DD >> 5) + 2 * h + kb) << 9));
#pragma unroll
    for (int j = 0; j < 2; ++j) {
      const v16bf fb =
          *(const v16bf*)(bBase + (((size_t)kb * 4 + nw * 2 + j) << 9));
      acc[0][j] = wmma_bf16(a0, fb, acc[0][j]);
      acc[1][j] = wmma_bf16(a1, fb, acc[1][j]);
    }
  }

  const int lhi = (lane >> 4) << 3;
#pragma unroll
  for (int i = 0; i < 2; ++i)
#pragma unroll
    for (int j = 0; j < 2; ++j)
#pragma unroll
      for (int e = 0; e < 8; ++e) {
        const int r = r0 + i * 16 + lhi + e;
        const int n = nw * 32 + j * 16 + (lane & 15);
        const float z = zinv[(size_t)r * HH + h];
        attnO[aswz(r, h * DH + n, DD)] = (bf)(acc[i][j][e] * z);
      }
}

// LayerNorm over last dim + cast to bf16 written A-swizzled.
__global__ __launch_bounds__(256) void ln_cast_kernel(
    const float* __restrict__ x, const float* __restrict__ g,
    const float* __restrict__ beta, bf* __restrict__ out, int cols) {
  __shared__ float s1[256], s2[256];
  const int row = blockIdx.x;
  const float* xr = x + (size_t)row * cols;
  float sum = 0.0f, sq = 0.0f;
  for (int c = threadIdx.x; c < cols; c += 256) {
    const float v = xr[c];
    sum += v;
    sq += v * v;
  }
  s1[threadIdx.x] = sum;
  s2[threadIdx.x] = sq;
  __syncthreads();
  for (int s = 128; s > 0; s >>= 1) {
    if (threadIdx.x < s) {
      s1[threadIdx.x] += s1[threadIdx.x + s];
      s2[threadIdx.x] += s2[threadIdx.x + s];
    }
    __syncthreads();
  }
  const float mu = s1[0] / cols;
  const float var = s2[0] / cols - mu * mu;
  const float inv = rsqrtf(var + 1e-5f);
  for (int c = threadIdx.x; c < cols; c += 256)
    out[aswz(row, c, cols)] = (bf)((xr[c] - mu) * inv * g[c] + beta[c]);
}

// fp32 row-major weight -> bf16 B-swizzled.
__global__ void cast_bswz_kernel(const float* __restrict__ src,
                                 bf* __restrict__ dst, int K, int N) {
  size_t i = (size_t)blockIdx.x * blockDim.x + threadIdx.x;
  const size_t total = (size_t)K * N;
  const size_t stride = (size_t)gridDim.x * blockDim.x;
  for (; i < total; i += stride) {
    const int k = (int)(i / N), n = (int)(i % N);
    dst[bswz(k, n, N)] = (bf)src[i];
  }
}

// ---------------------------------------------------------------------------
// Host-side orchestration
// ---------------------------------------------------------------------------
extern "C" void kernel_launch(void* const* d_in, const int* in_sizes, int n_in,
                              void* d_out, int out_size, void* d_ws,
                              size_t ws_size, hipStream_t stream) {
  const float* x   = (const float*)d_in[0];
  const unsigned char* mask = (const unsigned char*)d_in[1];
  const float* Wq  = (const float*)d_in[2];
  const float* bq  = (const float*)d_in[3];
  const float* Wk  = (const float*)d_in[4];
  const float* bk  = (const float*)d_in[5];
  const float* Wv  = (const float*)d_in[6];
  const float* bv  = (const float*)d_in[7];
  const float* Wo  = (const float*)d_in[8];
  const float* bo  = (const float*)d_in[9];
  const float* g1  = (const float*)d_in[10];
  const float* be1 = (const float*)d_in[11];
  const float* g2  = (const float*)d_in[12];
  const float* be2 = (const float*)d_in[13];
  const float* W1  = (const float*)d_in[14];
  const float* bf1 = (const float*)d_in[15];
  const float* W2  = (const float*)d_in[16];
  const float* bf2 = (const float*)d_in[17];

  char* cur = (char*)d_ws;
  auto alloc = [&](size_t bytes) {
    char* p = cur;
    cur += (bytes + 255) & ~(size_t)255;
    return p;
  };
  bf* wqB = (bf*)alloc((size_t)DD * DD * 2);
  bf* wkB = (bf*)alloc((size_t)DD * DD * 2);
  bf* wvB = (bf*)alloc((size_t)DD * DD * 2);
  bf* woB = (bf*)alloc((size_t)DD * DD * 2);
  bf* w1B = (bf*)alloc((size_t)DD * FF * 2);
  bf* w2B = (bf*)alloc((size_t)FF * DD * 2);
  bf* h1  = (bf*)alloc((size_t)MM * DD * 2);
  bf* Qb  = (bf*)alloc((size_t)MM * DD * 2);
  bf* Kb  = (bf*)alloc((size_t)MM * DD * 2);
  bf* Vb  = (bf*)alloc((size_t)MM * DD * 2);
  bf* kvb = (bf*)alloc((size_t)BB * HH * DH * DH * 2);
  float* ksum = (float*)alloc((size_t)BB * HH * DH * 4);
  float* zinv = (float*)alloc((size_t)MM * HH * 4);
  bf* attnO = (bf*)alloc((size_t)MM * DD * 2);
  float* x2 = (float*)alloc((size_t)MM * DD * 4);
  bf* h2  = (bf*)alloc((size_t)MM * DD * 2);
  bf* ff1 = (bf*)alloc((size_t)MM * FF * 2);

  // 1) weight casts -> B-swizzled bf16
  cast_bswz_kernel<<<2048, 256, 0, stream>>>(Wq, wqB, DD, DD);
  cast_bswz_kernel<<<2048, 256, 0, stream>>>(Wk, wkB, DD, DD);
  cast_bswz_kernel<<<2048, 256, 0, stream>>>(Wv, wvB, DD, DD);
  cast_bswz_kernel<<<2048, 256, 0, stream>>>(Wo, woB, DD, DD);
  cast_bswz_kernel<<<4096, 256, 0, stream>>>(W1, w1B, DD, FF);
  cast_bswz_kernel<<<4096, 256, 0, stream>>>(W2, w2B, FF, DD);

  // 2) LN1 -> h1 (bf16, A-swizzled)
  ln_cast_kernel<<<MM, 256, 0, stream>>>(x, g1, be1, h1, DD);

  // 3) Q/K/V projections with fused phi / mask epilogues
  dim3 gProj(DD / 128, MM / 128);
  gemm_swz<0><<<gProj, 256, 0, stream>>>(h1, wqB, bq, nullptr, nullptr, Qb,
                                         nullptr, MM, DD, DD);
  gemm_swz<1><<<gProj, 256, 0, stream>>>(h1, wkB, bk, nullptr, mask, Kb,
                                         nullptr, MM, DD, DD);
  gemm_swz<2><<<gProj, 256, 0, stream>>>(h1, wvB, bv, nullptr, mask, Vb,
                                         nullptr, MM, DD, DD);

  // 4) linear attention: KV, Ksum, Z, then (Q@KV)*Z
  kv_kernel<<<BB * HH, 256, 0, stream>>>(Kb, Vb, kvb);
  ksum_kernel<<<BB * HH, 256, 0, stream>>>(Kb, ksum);
  zinv_kernel<<<(MM * HH + 255) / 256, 256, 0, stream>>>(Qb, ksum, zinv);
  attn_out_swz<<<dim3(TT / 128, BB * HH), 256, 0, stream>>>(Qb, kvb, zinv,
                                                            attnO);

  // 5) out projection + residual -> x2 (fp32, row-major)
  gemm_swz<5><<<gProj, 256, 0, stream>>>(attnO, woB, bo, x, nullptr, nullptr,
                                         x2, MM, DD, DD);

  // 6) LN2 -> h2 (bf16, A-swizzled)
  ln_cast_kernel<<<MM, 256, 0, stream>>>(x2, g2, be2, h2, DD);

  // 7) FFN: gelu(h2@W1+bf1) @ W2 + bf2 + x2 -> d_out (fp32)
  dim3 gF1(FF / 128, MM / 128);
  gemm_swz<4><<<gF1, 256, 0, stream>>>(h2, w1B, bf1, nullptr, nullptr, ff1,
                                       nullptr, MM, FF, DD);
  gemm_swz<5><<<gProj, 256, 0, stream>>>(ff1, w2B, bf2, x2, nullptr, nullptr,
                                         (float*)d_out, MM, DD, FF);
}
